// ResBlock_3d_31044023616354
// MI455X (gfx1250) — compile-verified, matrix operand order checked by reading, not run on hardware
//
#include <hip/hip_runtime.h>
#include <hip/hip_bf16.h>

// ---------------- types for WMMA fragments ----------------
typedef _Float16 v16h __attribute__((ext_vector_type(16)));
typedef _Float16 h8v  __attribute__((ext_vector_type(8)));
typedef float    v8f  __attribute__((ext_vector_type(8)));

// ---------------- problem constants ----------------
constexpr int Tn  = 4;
constexpr int Hn  = 96;
constexpr int Wn  = 96;
constexpr int Cn  = 64;
constexpr int HWn = Hn * Wn;              // 9216
constexpr int Pn  = Tn * HWn;             // 36864 output positions
constexpr int Ktot = 27 * Cn;             // 1728 GEMM K dimension
constexpr int LDSK = 72;                  // padded A row stride (144B, conflict-free b128)

// =========================================================
// Prep: fp32 OIDHW conv weight [Nsrc][64][27] -> f16 column-major [64][1728]
// column n, k = tap*64 + cin ; rows >= Nsrc are zero padding.
// =========================================================
__global__ __launch_bounds__(256) void prep_weight(const float* __restrict__ src,
                                                   _Float16* __restrict__ dst,
                                                   int Nsrc) {
    int idx = blockIdx.x * 256 + threadIdx.x;
    if (idx >= Cn * Ktot) return;
    int n   = idx / Ktot;
    int kk  = idx - n * Ktot;
    int tap = kk >> 6;           // 0..26
    int cin = kk & 63;
    float v = (n < Nsrc) ? src[((size_t)n * Cn + cin) * 27 + tap] : 0.0f;
    dst[(size_t)n * Ktot + kk] = (_Float16)v;
}

// =========================================================
// Prep: x fp32 NCDHW -> f16 channels-last [p][64]
// coalesced reads (consecutive p), per-thread 128B contiguous writes.
// =========================================================
__global__ __launch_bounds__(256) void x_to_chlast(const float* __restrict__ x,
                                                   _Float16* __restrict__ xh) {
    int p = blockIdx.x * 256 + threadIdx.x;
    if (p >= Pn) return;
    _Float16 tmp[Cn];
#pragma unroll
    for (int c = 0; c < Cn; ++c) tmp[c] = (_Float16)x[(size_t)c * Pn + p];
    h8v* dp = (h8v*)(xh + (size_t)p * Cn);
    const h8v* sp = (const h8v*)tmp;
#pragma unroll
    for (int i = 0; i < 8; ++i) dp[i] = sp[i];
}

// =========================================================
// Implicit-GEMM DCN kernel.
//   SMODE 0: A-rows from integer 3x3x3 taps (offset-predicting conv)
//   SMODE 1: A-rows from bilinear deformable sampling using `off`
//   EMODE 0: out = acc + b_off          -> fp32 [p][64]   (offsets)
//   EMODE 1: out = leaky(acc + b, .1)   -> f16 ch-last    (layer-0 act)
//   EMODE 2: out = acc + b + x_resid    -> fp32 NCDHW d_out
// Block: 256 thr (8 waves), 64 positions x 64 out-channels tile.
// =========================================================
template <int SMODE, int EMODE>
__global__ __launch_bounds__(256) void dcn_gemm(
    const _Float16* __restrict__ src,   // channels-last f16 [P][64] sampling source
    const float*    __restrict__ off,   // [P][64] fp32 (dh,dw per tap) ; SMODE==1
    const _Float16* __restrict__ Bmat,  // f16 column-major [64][1728]
    const float*    __restrict__ bias,
    float*          __restrict__ offOut,
    _Float16*       __restrict__ hOut,
    float*          __restrict__ yOut,
    const float*    __restrict__ xres) {
    __shared__ _Float16 Alds[64 * LDSK];

    const int pBase = blockIdx.x * 64;
    const int tid   = threadIdx.x;
    const int wave  = tid >> 5;
    const int lane  = tid & 31;
    const int l15   = lane & 15;
    const int hi    = lane >> 4;       // 0/1: K-halves (A/B), M+8 (C/D)
    const int mT0   = (wave >> 2) * 2; // wave -> two M-subtiles, one N-subtile
    const int nT    = wave & 3;

    // sampling mapping: 4 threads per position, 16 channels each
    const int pl  = tid >> 2;
    const int cg  = tid & 3;
    const int p   = pBase + pl;
    const int t0  = p / HWn;
    const int rem = p - t0 * HWn;
    const int h0c = rem / Wn;
    const int w0c = rem - h0c * Wn;

    v8f acc0 = {};
    v8f acc1 = {};

    const _Float16* bcol = Bmat + (size_t)(nT * 16 + l15) * Ktot;

    for (int k = 0; k < 27; ++k) {
        const int kt = k / 9 - 1;
        const int kh = (k / 3) % 3 - 1;
        const int kw = k % 3 - 1;

        // ---------- build A chunk for this tap: 64 rows x 64 (cin) f16 ----------
        float v[16];
#pragma unroll
        for (int i = 0; i < 16; ++i) v[i] = 0.0f;

        if constexpr (SMODE == 0) {
            const int tt = t0 + kt, hh = h0c + kh, ww = w0c + kw;
            const bool ok = (tt >= 0) & (tt < Tn) & (hh >= 0) & (hh < Hn) &
                            (ww >= 0) & (ww < Wn);
            if (ok) {
                const _Float16* sp =
                    src + ((size_t)(tt * HWn + hh * Wn + ww)) * Cn + cg * 16;
                const h8v a0 = *(const h8v*)sp;
                const h8v a1 = *(const h8v*)(sp + 8);
#pragma unroll
                for (int i = 0; i < 8; ++i) {
                    v[i]     = (float)a0[i];
                    v[i + 8] = (float)a1[i];
                }
            }
        } else {
            const int  tt  = t0 + kt;
            const bool tok = (tt >= 0) && (tt < Tn);
            const int  tc  = tt < 0 ? 0 : (tt >= Tn ? Tn - 1 : tt);
            const float dh = off[(size_t)p * 64 + 2 * k];
            const float dw = off[(size_t)p * 64 + 2 * k + 1];
            const float hp = (float)(h0c + kh) + dh;
            const float wp = (float)(w0c + kw) + dw;
            const float hf = floorf(hp), wf = floorf(wp);
            const float ah = hp - hf, aw = wp - wf;
            const int h0i = (int)hf, w0i = (int)wf;
            const float wgt[4] = {(1.f - ah) * (1.f - aw), (1.f - ah) * aw,
                                  ah * (1.f - aw), ah * aw};
            const int hc[4] = {h0i, h0i, h0i + 1, h0i + 1};
            const int wc[4] = {w0i, w0i + 1, w0i, w0i + 1};
#pragma unroll
            for (int c4 = 0; c4 < 4; ++c4) {
                const bool ok = tok && hc[c4] >= 0 && hc[c4] < Hn &&
                                wc[c4] >= 0 && wc[c4] < Wn;
                if (ok) {
                    const _Float16* sp =
                        src + ((size_t)(tc * HWn + hc[c4] * Wn + wc[c4])) * Cn +
                        cg * 16;
                    const h8v a0 = *(const h8v*)sp;
                    const h8v a1 = *(const h8v*)(sp + 8);
                    const float g = wgt[c4];
#pragma unroll
                    for (int i = 0; i < 8; ++i) {
                        v[i]     += g * (float)a0[i];
                        v[i + 8] += g * (float)a1[i];
                    }
                }
            }
        }
        {
            h8v s0, s1;
#pragma unroll
            for (int i = 0; i < 8; ++i) {
                s0[i] = (_Float16)v[i];
                s1[i] = (_Float16)v[i + 8];
            }
            _Float16* ap = &Alds[pl * LDSK + cg * 16];
            *(h8v*)ap       = s0;
            *(h8v*)(ap + 8) = s1;
        }
        __syncthreads();

        // ---------- WMMA: two K=32 steps over this 64-wide chunk ----------
        const int kGlob = k * 64;
#pragma unroll
        for (int ks = 0; ks < 2; ++ks) {
            // B fragment: lane = column N ; hi half takes K+8 (ISA 16-bit layout)
            union { v16h v; h8v h[2]; } ub, ua0, ua1;
            const _Float16* bp = bcol + kGlob + ks * 32 + hi * 8;
            ub.h[0] = *(const h8v*)bp;
            ub.h[1] = *(const h8v*)(bp + 16);
            const _Float16* a0p =
                &Alds[(mT0 * 16 + l15) * LDSK + ks * 32 + hi * 8];
            ua0.h[0] = *(const h8v*)a0p;
            ua0.h[1] = *(const h8v*)(a0p + 16);
            const _Float16* a1p =
                &Alds[((mT0 + 1) * 16 + l15) * LDSK + ks * 32 + hi * 8];
            ua1.h[0] = *(const h8v*)a1p;
            ua1.h[1] = *(const h8v*)(a1p + 16);

            acc0 = __builtin_amdgcn_wmma_f32_16x16x32_f16(
                false, ua0.v, false, ub.v, (short)0, acc0, false, false);
            acc1 = __builtin_amdgcn_wmma_f32_16x16x32_f16(
                false, ua1.v, false, ub.v, (short)0, acc1, false, false);
        }
        __syncthreads();
    }

    // ---------- epilogue (C/D layout: lane = N, VGPR v -> M = v + 8*hi) ----------
    const int n = nT * 16 + l15;
    float bn;
    if constexpr (EMODE == 0)
        bn = (n < 54) ? bias[n] : 0.0f;
    else
        bn = bias[n];

#pragma unroll
    for (int tile = 0; tile < 2; ++tile) {
        const v8f a = tile ? acc1 : acc0;
        const int mBase = (mT0 + tile) * 16 + hi * 8;
#pragma unroll
        for (int vv = 0; vv < 8; ++vv) {
            const int pp = pBase + mBase + vv;
            float val = a[vv] + bn;
            if constexpr (EMODE == 0) {
                offOut[(size_t)pp * 64 + n] = val;
            } else if constexpr (EMODE == 1) {
                val = val >= 0.0f ? val : 0.1f * val;
                hOut[(size_t)pp * Cn + n] = (_Float16)val;
            } else {
                yOut[(size_t)n * Pn + pp] = val + xres[(size_t)n * Pn + pp];
            }
        }
    }
}

// =========================================================
// Host side
// =========================================================
extern "C" void kernel_launch(void* const* d_in, const int* in_sizes, int n_in,
                              void* d_out, int out_size, void* d_ws, size_t ws_size,
                              hipStream_t stream) {
    (void)in_sizes; (void)n_in; (void)out_size; (void)ws_size;
    const float* x      = (const float*)d_in[0];
    const float* w_off0 = (const float*)d_in[1];
    const float* b_off0 = (const float*)d_in[2];
    const float* w0     = (const float*)d_in[3];
    const float* b0     = (const float*)d_in[4];
    const float* w_off1 = (const float*)d_in[5];
    const float* b_off1 = (const float*)d_in[6];
    const float* w1     = (const float*)d_in[7];
    const float* b1     = (const float*)d_in[8];
    float* out = (float*)d_out;

    // workspace layout (all sizes multiples of 256B)
    char* ws = (char*)d_ws;
    const size_t szXH  = (size_t)Pn * Cn * sizeof(_Float16);   // 4,718,592
    const size_t szOFF = (size_t)Pn * 64 * sizeof(float);      // 9,437,184
    const size_t szB   = (size_t)Cn * Ktot * sizeof(_Float16); //   221,184
    _Float16* xh    = (_Float16*)(ws);
    _Float16* hh    = (_Float16*)(ws + szXH);
    float*    offb  = (float*)   (ws + 2 * szXH);
    _Float16* B0off = (_Float16*)(ws + 2 * szXH + szOFF);
    _Float16* B0mn  = (_Float16*)(ws + 2 * szXH + szOFF + szB);
    _Float16* B1off = (_Float16*)(ws + 2 * szXH + szOFF + 2 * szB);
    _Float16* B1mn  = (_Float16*)(ws + 2 * szXH + szOFF + 3 * szB);

    const int wElems = Cn * Ktot;                 // 110592
    const int wBlk   = (wElems + 255) / 256;      // 432
    prep_weight<<<wBlk, 256, 0, stream>>>(w_off0, B0off, 54);
    prep_weight<<<wBlk, 256, 0, stream>>>(w0,     B0mn,  64);
    prep_weight<<<wBlk, 256, 0, stream>>>(w_off1, B1off, 54);
    prep_weight<<<wBlk, 256, 0, stream>>>(w1,     B1mn,  64);
    x_to_chlast<<<Pn / 256, 256, 0, stream>>>(x, xh);

    const int gBlk = Pn / 64;                     // 576

    // layer 0: offsets, then deformable GEMM with LeakyReLU -> hh (f16 ch-last)
    dcn_gemm<0, 0><<<gBlk, 256, 0, stream>>>(xh, nullptr, B0off, b_off0,
                                             offb, nullptr, nullptr, nullptr);
    dcn_gemm<1, 1><<<gBlk, 256, 0, stream>>>(xh, offb, B0mn, b0,
                                             nullptr, hh, nullptr, nullptr);
    // layer 1: offsets from hh, deformable GEMM + residual -> d_out (fp32 NCDHW)
    dcn_gemm<0, 0><<<gBlk, 256, 0, stream>>>(hh, nullptr, B1off, b_off1,
                                             offb, nullptr, nullptr, nullptr);
    dcn_gemm<1, 2><<<gBlk, 256, 0, stream>>>(hh, offb, B1mn, b1,
                                             nullptr, nullptr, out, x);
}
